// FFM_23167053595302
// MI455X (gfx1250) — matrix operand order repaired and checked by reading, not compiled, not run on hardware
//
#include <hip/hip_runtime.h>
#include <hip/hip_bf16.h>
#include <math.h>

// Problem constants (match reference)
#define T_STEPS 4096
#define DIN     512
#define TRACE   64
#define CTX     64
#define OUTD    512
#define PAIRS   (TRACE * CTX)       // 4096 independent complex modes
#define ZIND    (2 * TRACE * CTX)   // 8192 = z_in feature dim
#define NCHUNK  32
#define CLEN    (T_STEPS / NCHUNK)  // 128
#define LN_EPS  1e-6f

typedef __bf16 bf16;
typedef __attribute__((ext_vector_type(16))) __bf16 v16bf;
typedef __attribute__((ext_vector_type(8)))  __bf16 v8bf;
typedef __attribute__((ext_vector_type(8)))  float  v8f;

union Frag { v16bf v; v8bf h[2]; };

// ---------------------------------------------------------------------------
// Elementwise converters
// ---------------------------------------------------------------------------
__global__ void f32_to_bf16_kernel(const float* __restrict__ src,
                                   bf16* __restrict__ dst, int n) {
  int tid = blockIdx.x * blockDim.x + threadIdx.x;
  if (tid < n) dst[tid] = (bf16)src[tid];
}

// W is (K x N) f32 row-major; WT is (N x K) bf16 row-major (for WMMA B frags)
__global__ void transpose_f32_to_bf16(const float* __restrict__ W,
                                      bf16* __restrict__ WT, int K, int N) {
  int tid = blockIdx.x * blockDim.x + threadIdx.x;
  if (tid >= K * N) return;
  int n = tid / K;
  int k = tid - n * K;
  WT[(size_t)n * K + k] = (bf16)W[(size_t)k * N + n];
}

__global__ void gated_kernel(const float* __restrict__ pre,
                             const float* __restrict__ gi,
                             float* __restrict__ gated, int n) {
  int tid = blockIdx.x * blockDim.x + threadIdx.x;
  if (tid < n) gated[tid] = pre[tid] * (1.f / (1.f + __expf(-gi[tid])));
}

// ---------------------------------------------------------------------------
// bf16 WMMA GEMM: C(MxN) = A(MxK) @ B(KxN) + bias, A bf16 row-major,
// BT bf16 = B^T row-major (N x K), C f32.
// Wave tile 64x32 (4 m-tiles x 2 n-tiles -> 8 WMMA accumulators); 2x2 waves
// per block -> 128x64 block tile. 12 b128 loads feed 8 WMMAs per k-step
// (1.5 loads/WMMA) to cut L2 operand traffic. M mult of 128, N of 64, K of 32.
// ---------------------------------------------------------------------------
__global__ __launch_bounds__(128)
void gemm_bf16_wmma(const bf16* __restrict__ A, const bf16* __restrict__ BT,
                    const float* __restrict__ bias, float* __restrict__ C,
                    int M, int N, int K) {
  const int lane  = threadIdx.x & 31;
  const int wave  = threadIdx.x >> 5;
  const int l15   = lane & 15;
  const int lhalf = lane >> 4;
  const int m0 = blockIdx.x * 128 + (wave >> 1) * 64;
  const int n0 = blockIdx.y * 64  + (wave & 1) * 32;

  // A frag (16x32 bf16): lane<16 holds row M=l15, K={0..7,16..23}; lane>=16 +8
  const bf16* ap0 = A + (size_t)(m0 + l15) * K + lhalf * 8;
  const bf16* ap1 = ap0 + (size_t)16 * K;
  const bf16* ap2 = ap0 + (size_t)32 * K;
  const bf16* ap3 = ap0 + (size_t)48 * K;
  // B frag (32x16 bf16): lane holds col N=l15, K={0..7,16..23}; lane>=16 +8
  const bf16* bp0 = BT + (size_t)(n0 + l15) * K + lhalf * 8;
  const bf16* bp1 = bp0 + (size_t)16 * K;

  v8f acc[4][2] = {};
  for (int k0 = 0; k0 < K; k0 += 32) {
    Frag a0, a1, a2, a3, b0, b1;
    a0.h[0] = *(const v8bf*)(ap0 + k0);
    a0.h[1] = *(const v8bf*)(ap0 + k0 + 16);
    a1.h[0] = *(const v8bf*)(ap1 + k0);
    a1.h[1] = *(const v8bf*)(ap1 + k0 + 16);
    a2.h[0] = *(const v8bf*)(ap2 + k0);
    a2.h[1] = *(const v8bf*)(ap2 + k0 + 16);
    a3.h[0] = *(const v8bf*)(ap3 + k0);
    a3.h[1] = *(const v8bf*)(ap3 + k0 + 16);
    b0.h[0] = *(const v8bf*)(bp0 + k0);
    b0.h[1] = *(const v8bf*)(bp0 + k0 + 16);
    b1.h[0] = *(const v8bf*)(bp1 + k0);
    b1.h[1] = *(const v8bf*)(bp1 + k0 + 16);
    acc[0][0] = __builtin_amdgcn_wmma_f32_16x16x32_bf16(false, a0.v, false, b0.v,
                                                        (short)0, acc[0][0], false, false);
    acc[0][1] = __builtin_amdgcn_wmma_f32_16x16x32_bf16(false, a0.v, false, b1.v,
                                                        (short)0, acc[0][1], false, false);
    acc[1][0] = __builtin_amdgcn_wmma_f32_16x16x32_bf16(false, a1.v, false, b0.v,
                                                        (short)0, acc[1][0], false, false);
    acc[1][1] = __builtin_amdgcn_wmma_f32_16x16x32_bf16(false, a1.v, false, b1.v,
                                                        (short)0, acc[1][1], false, false);
    acc[2][0] = __builtin_amdgcn_wmma_f32_16x16x32_bf16(false, a2.v, false, b0.v,
                                                        (short)0, acc[2][0], false, false);
    acc[2][1] = __builtin_amdgcn_wmma_f32_16x16x32_bf16(false, a2.v, false, b1.v,
                                                        (short)0, acc[2][1], false, false);
    acc[3][0] = __builtin_amdgcn_wmma_f32_16x16x32_bf16(false, a3.v, false, b0.v,
                                                        (short)0, acc[3][0], false, false);
    acc[3][1] = __builtin_amdgcn_wmma_f32_16x16x32_bf16(false, a3.v, false, b1.v,
                                                        (short)0, acc[3][1], false, false);
  }

  // C/D layout: VGPR r, lanes 0..15 -> M=r, lanes 16..31 -> M=r+8; N = l15
#pragma unroll
  for (int mt = 0; mt < 4; ++mt) {
#pragma unroll
    for (int nt = 0; nt < 2; ++nt) {
      int col = n0 + nt * 16 + l15;
      float bb = bias[col];
#pragma unroll
      for (int r = 0; r < 8; ++r) {
        int row = m0 + mt * 16 + r + lhalf * 8;
        C[(size_t)row * N + col] = acc[mt][nt][r] + bb;
      }
    }
  }
}

// ---------------------------------------------------------------------------
// Segmented complex linear scan, chunk-parallel (3 phases).
// Mode p = i*64 + k; lambda = exp(-|a_i|) * (cos b_k + j sin b_k).
// s[t] = resets[t] ? z[t] : s[t-1]*lambda + z[t],  z[t] = gated[t,i] (real).
// ---------------------------------------------------------------------------
__global__ void scan_chunk_reduce(const float* __restrict__ gated,
                                  const unsigned char* __restrict__ resets,
                                  const float* __restrict__ a,
                                  const float* __restrict__ b,
                                  float* __restrict__ chA_re, float* __restrict__ chA_im,
                                  float* __restrict__ chB_re, float* __restrict__ chB_im) {
  int gid = blockIdx.x * blockDim.x + threadIdx.x;  // NCHUNK*PAIRS
  int c = gid >> 12;            // / PAIRS
  int p = gid & (PAIRS - 1);
  int i = p >> 6, k = p & 63;
  float decay = __expf(-fabsf(a[i]));
  float lr = decay * __cosf(b[k]);
  float li = decay * __sinf(b[k]);
  float Ar = 1.f, Ai = 0.f, Br = 0.f, Bi = 0.f;  // s_end = A*s_in + B
  int t0 = c * CLEN;
  for (int t = t0; t < t0 + CLEN; ++t) {
    float z = gated[t * TRACE + i];
    if (resets[t]) {
      Ar = 0.f; Ai = 0.f; Br = z; Bi = 0.f;
    } else {
      float nAr = Ar * lr - Ai * li; Ai = Ar * li + Ai * lr; Ar = nAr;
      float nBr = Br * lr - Bi * li + z; Bi = Br * li + Bi * lr; Br = nBr;
    }
  }
  chA_re[gid] = Ar; chA_im[gid] = Ai; chB_re[gid] = Br; chB_im[gid] = Bi;
}

__global__ void scan_chunk_combine(const float* __restrict__ state_re,
                                   const float* __restrict__ state_im,
                                   const float* __restrict__ chA_re,
                                   const float* __restrict__ chA_im,
                                   const float* __restrict__ chB_re,
                                   const float* __restrict__ chB_im,
                                   float* __restrict__ sIn_re,
                                   float* __restrict__ sIn_im) {
  int p = blockIdx.x * blockDim.x + threadIdx.x;  // PAIRS
  float sr = state_re[p], si = state_im[p];
  for (int c = 0; c < NCHUNK; ++c) {
    int idx = c * PAIRS + p;
    sIn_re[idx] = sr; sIn_im[idx] = si;
    float Ar = chA_re[idx], Ai = chA_im[idx];
    float Br = chB_re[idx], Bi = chB_im[idx];
    float nr = Ar * sr - Ai * si + Br;
    si = Ar * si + Ai * sr + Bi;
    sr = nr;
  }
}

__global__ void scan_chunk_apply(const float* __restrict__ gated,
                                 const unsigned char* __restrict__ resets,
                                 const float* __restrict__ a,
                                 const float* __restrict__ b,
                                 const float* __restrict__ sIn_re,
                                 const float* __restrict__ sIn_im,
                                 bf16* __restrict__ zin,
                                 float* __restrict__ final_state) {
  int gid = blockIdx.x * blockDim.x + threadIdx.x;
  int c = gid >> 12;
  int p = gid & (PAIRS - 1);
  int i = p >> 6, k = p & 63;
  float decay = __expf(-fabsf(a[i]));
  float lr = decay * __cosf(b[k]);
  float li = decay * __sinf(b[k]);
  float sr = sIn_re[gid], si = sIn_im[gid];
  int t0 = c * CLEN;
  for (int t = t0; t < t0 + CLEN; ++t) {
    float z = gated[t * TRACE + i];
    if (resets[t]) {
      sr = z; si = 0.f;
    } else {
      float nr = sr * lr - si * li + z;
      si = sr * li + si * lr;
      sr = nr;
    }
    size_t base = (size_t)t * ZIND + (size_t)i * (2 * CTX) + k;
    zin[base]       = (bf16)sr;   // real half
    zin[base + CTX] = (bf16)si;   // imag half
  }
  if (c == NCHUNK - 1) {  // rs[-1] -> complex64 interleaved re/im
    final_state[2 * p]     = sr;
    final_state[2 * p + 1] = si;
  }
}

// ---------------------------------------------------------------------------
// Fused epilogue: v = zm*sigmoid(go); LN(v) + skip*(1-sigmoid(go)).
// One wave per row (512 cols -> 16 per lane), __shfl_xor wave32 reduction.
// ---------------------------------------------------------------------------
__global__ __launch_bounds__(256)
void fused_gate_ln(const float* __restrict__ zm, const float* __restrict__ go_raw,
                   const float* __restrict__ skip, float* __restrict__ out) {
  int lane = threadIdx.x & 31;
  int row  = blockIdx.x * 8 + (threadIdx.x >> 5);
  const float* zr = zm     + (size_t)row * OUTD;
  const float* gr = go_raw + (size_t)row * OUTD;
  const float* sr = skip   + (size_t)row * OUTD;
  float v[16], g[16], sk[16];
  float sum = 0.f, sumsq = 0.f;
#pragma unroll
  for (int j = 0; j < 16; ++j) {
    int col = lane + j * 32;
    float gg = 1.f / (1.f + __expf(-gr[col]));
    float vv = zr[col] * gg;
    v[j] = vv; g[j] = gg; sk[j] = sr[col];
    sum += vv; sumsq += vv * vv;
  }
#pragma unroll
  for (int off = 16; off >= 1; off >>= 1) {
    sum   += __shfl_xor(sum, off, 32);
    sumsq += __shfl_xor(sumsq, off, 32);
  }
  float mu   = sum * (1.f / OUTD);
  float var  = sumsq * (1.f / OUTD) - mu * mu;
  float rstd = rsqrtf(var + LN_EPS);
  float* orow = out + (size_t)row * OUTD;
#pragma unroll
  for (int j = 0; j < 16; ++j) {
    int col = lane + j * 32;
    orow[col] = (v[j] - mu) * rstd + sk[j] * (1.f - g[j]);
  }
}

// ---------------------------------------------------------------------------
extern "C" void kernel_launch(void* const* d_in, const int* in_sizes, int n_in,
                              void* d_out, int out_size, void* d_ws, size_t ws_size,
                              hipStream_t stream) {
  const float* x        = (const float*)d_in[0];
  const unsigned char* resets = (const unsigned char*)d_in[1];  // jnp bool -> 1 byte
  const float* state_re = (const float*)d_in[2];
  const float* state_im = (const float*)d_in[3];
  const float* a        = (const float*)d_in[4];
  const float* b        = (const float*)d_in[5];
  const float* W_pre  = (const float*)d_in[6];
  const float* b_pre  = (const float*)d_in[7];
  const float* W_gi   = (const float*)d_in[8];
  const float* b_gi   = (const float*)d_in[9];
  const float* W_go   = (const float*)d_in[10];
  const float* b_go   = (const float*)d_in[11];
  const float* W_skip = (const float*)d_in[12];
  const float* b_skip = (const float*)d_in[13];
  const float* W_mix  = (const float*)d_in[14];
  const float* b_mix  = (const float*)d_in[15];

  char* wsp = (char*)d_ws;
  auto alloc = [&](size_t bytes) -> char* {
    char* p = wsp;
    wsp += (bytes + 255) & ~(size_t)255;
    return p;
  };

  bf16*  xb      = (bf16*)alloc((size_t)T_STEPS * DIN * 2);
  bf16*  WpreT   = (bf16*)alloc((size_t)TRACE * DIN * 2);
  bf16*  WgiT    = (bf16*)alloc((size_t)TRACE * DIN * 2);
  bf16*  WgoT    = (bf16*)alloc((size_t)OUTD * DIN * 2);
  bf16*  WskipT  = (bf16*)alloc((size_t)OUTD * DIN * 2);
  bf16*  WmixT   = (bf16*)alloc((size_t)OUTD * ZIND * 2);
  float* pre_raw = (float*)alloc((size_t)T_STEPS * TRACE * 4);
  float* gi_raw  = (float*)alloc((size_t)T_STEPS * TRACE * 4);
  float* gated   = (float*)alloc((size_t)T_STEPS * TRACE * 4);
  float* chA_re  = (float*)alloc((size_t)NCHUNK * PAIRS * 4);
  float* chA_im  = (float*)alloc((size_t)NCHUNK * PAIRS * 4);
  float* chB_re  = (float*)alloc((size_t)NCHUNK * PAIRS * 4);
  float* chB_im  = (float*)alloc((size_t)NCHUNK * PAIRS * 4);
  float* sIn_re  = (float*)alloc((size_t)NCHUNK * PAIRS * 4);
  float* sIn_im  = (float*)alloc((size_t)NCHUNK * PAIRS * 4);
  bf16*  zin     = (bf16*)alloc((size_t)T_STEPS * ZIND * 2);
  float* zm      = (float*)alloc((size_t)T_STEPS * OUTD * 4);
  float* go_raw  = (float*)alloc((size_t)T_STEPS * OUTD * 4);
  float* skp     = (float*)alloc((size_t)T_STEPS * OUTD * 4);

  float* out_fs  = (float*)d_out;               // 8192 floats: complex64 final_state
  float* out_seq = (float*)d_out + 2 * PAIRS;   // T x 512 output

  // 1) bf16 copies / transposes of operands
  f32_to_bf16_kernel<<<(T_STEPS * DIN) / 256, 256, 0, stream>>>(x, xb, T_STEPS * DIN);
  transpose_f32_to_bf16<<<(DIN * TRACE + 255) / 256, 256, 0, stream>>>(W_pre, WpreT, DIN, TRACE);
  transpose_f32_to_bf16<<<(DIN * TRACE + 255) / 256, 256, 0, stream>>>(W_gi, WgiT, DIN, TRACE);
  transpose_f32_to_bf16<<<(DIN * OUTD + 255) / 256, 256, 0, stream>>>(W_go, WgoT, DIN, OUTD);
  transpose_f32_to_bf16<<<(DIN * OUTD + 255) / 256, 256, 0, stream>>>(W_skip, WskipT, DIN, OUTD);
  transpose_f32_to_bf16<<<(ZIND * OUTD + 255) / 256, 256, 0, stream>>>(W_mix, WmixT, ZIND, OUTD);

  // 2) input-side GEMMs (WMMA bf16); block tile 128x64
  gemm_bf16_wmma<<<dim3(T_STEPS / 128, TRACE / 64), 128, 0, stream>>>(
      xb, WpreT, b_pre, pre_raw, T_STEPS, TRACE, DIN);
  gemm_bf16_wmma<<<dim3(T_STEPS / 128, TRACE / 64), 128, 0, stream>>>(
      xb, WgiT, b_gi, gi_raw, T_STEPS, TRACE, DIN);
  gated_kernel<<<(T_STEPS * TRACE) / 256, 256, 0, stream>>>(
      pre_raw, gi_raw, gated, T_STEPS * TRACE);

  // 3) chunk-parallel segmented complex scan -> z_in (bf16) + final_state
  scan_chunk_reduce<<<(NCHUNK * PAIRS) / 256, 256, 0, stream>>>(
      gated, resets, a, b, chA_re, chA_im, chB_re, chB_im);
  scan_chunk_combine<<<PAIRS / 256, 256, 0, stream>>>(
      state_re, state_im, chA_re, chA_im, chB_re, chB_im, sIn_re, sIn_im);
  scan_chunk_apply<<<(NCHUNK * PAIRS) / 256, 256, 0, stream>>>(
      gated, resets, a, b, sIn_re, sIn_im, zin, out_fs);

  // 4) dominant GEMM (z_in @ W_mix) + output-side GEMMs
  gemm_bf16_wmma<<<dim3(T_STEPS / 128, OUTD / 64), 128, 0, stream>>>(
      zin, WmixT, b_mix, zm, T_STEPS, OUTD, ZIND);
  gemm_bf16_wmma<<<dim3(T_STEPS / 128, OUTD / 64), 128, 0, stream>>>(
      xb, WgoT, b_go, go_raw, T_STEPS, OUTD, DIN);
  gemm_bf16_wmma<<<dim3(T_STEPS / 128, OUTD / 64), 128, 0, stream>>>(
      xb, WskipT, b_skip, skp, T_STEPS, OUTD, DIN);

  // 5) fused gate + LayerNorm + skip epilogue
  fused_gate_ln<<<T_STEPS / 8, 256, 0, stream>>>(zm, go_raw, skp, out_seq);
}
